// SampledSoftmaxLoss_58162447123212
// MI455X (gfx1250) — matrix operand order repaired and checked by reading, not compiled, not run on hardware
//
#include <hip/hip_runtime.h>
#include <hip/hip_bf16.h>
#include <math.h>

typedef __attribute__((ext_vector_type(16))) _Float16 v16h;
typedef __attribute__((ext_vector_type(8)))  _Float16 v8h;
typedef __attribute__((ext_vector_type(4)))  _Float16 v4h;
typedef __attribute__((ext_vector_type(8)))  float    v8f;
typedef __attribute__((ext_vector_type(4)))  float    v4f;

#define D_DIM        128
#define KNEG         256
#define ROWS_PER_BLK 8
#define ESTRIDE      144    // halves per candidate row in LDS (128 + 16 pad, keeps 32B align)
#define INV_TEMP     20.0f  // 1 / 0.05
#define EPSN         1e-12f

__device__ __forceinline__ float wave_sum32(float v) {
#pragma unroll
  for (int off = 16; off >= 1; off >>= 1)
    v += __shfl_xor(v, off, 32);
  return v;
}

__global__ void __launch_bounds__(ROWS_PER_BLK * 32)
ssl_row_kernel(const float* __restrict__ out_emb,
               const float* __restrict__ table,
               const float* __restrict__ sw,
               const int*   __restrict__ tgt,
               const int*   __restrict__ neg,
               float* __restrict__ lw,     // [n_rows] weighted per-row loss
               int n_rows, int table_rows)
{
  // wave-private LDS slices: no inter-wave sync needed anywhere
  __shared__ __align__(32) _Float16 sQ[ROWS_PER_BLK][D_DIM];
  __shared__ __align__(32) _Float16 sE[ROWS_PER_BLK][16 * ESTRIDE];

  const int wave = threadIdx.x >> 5;
  const int lane = threadIdx.x & 31;
  const int n = blockIdx.x * ROWS_PER_BLK + wave;
  if (n >= n_rows) return;                 // wave-uniform

  const float w = sw[n];                   // same value across wave -> uniform
  if (w == 0.0f) { if (lane == 0) lw[n] = 0.0f; return; }

  const int maxidx = table_rows - 1;

  // ---- normalize q (f32) ----
  v4f qv = *(const v4f*)&out_emb[(size_t)n * D_DIM + lane * 4];
  float qss = wave_sum32(qv[0]*qv[0] + qv[1]*qv[1] + qv[2]*qv[2] + qv[3]*qv[3]);
  float qsc = 1.0f / fmaxf(sqrtf(qss), EPSN);
  qv *= qsc;

  // ---- positive logit in full f32 ----
  int t = tgt[n] - 1;
  t = t < 0 ? 0 : (t > maxidx ? maxidx : t);
  v4f pv = *(const v4f*)&table[(size_t)t * D_DIM + lane * 4];
  float pss  = wave_sum32(pv[0]*pv[0] + pv[1]*pv[1] + pv[2]*pv[2] + pv[3]*pv[3]);
  float psc  = 1.0f / fmaxf(sqrtf(pss), EPSN);
  float pdot = wave_sum32(qv[0]*pv[0] + qv[1]*pv[1] + qv[2]*pv[2] + qv[3]*pv[3]);
  float pos_t = pdot * psc * INV_TEMP;

  // ---- stage q-hat as f16, build replicated-A fragments (contiguous K order) ----
#pragma unroll
  for (int k = 0; k < 4; ++k) sQ[wave][lane * 4 + k] = (_Float16)qv[k];

  const int ahalf = (lane & 16) ? 16 : 0;  // lanes 16-31 hold K+16 (contiguous K order, matches B)
  v16h Afrag[4];
#pragma unroll
  for (int ck = 0; ck < 4; ++ck) {
    v8h lo8 = *(const v8h*)&sQ[wave][ck * 32 + ahalf];        // K = ck*32 + {0..7 | 16..23}
    v8h hi8 = *(const v8h*)&sQ[wave][ck * 32 + 8 + ahalf];    // K = ck*32 + {8..15 | 24..31}
    v16h a;
#pragma unroll
    for (int i = 0; i < 8; ++i) { a[i] = lo8[i]; a[i + 8] = hi8[i]; }
    Afrag[ck] = a;
  }

  float m = -__builtin_inff(), s = 0.0f;   // online log-sum-exp state (lanes 0-15)
  const int col = lane & 15;
  const int bko = (lane & 16) ? 16 : 0;    // B layout: lanes 0-15 K=0..15, lanes 16-31 K=16..31

  for (int c = 0; c < KNEG / 16; ++c) {
    int myid = 0;
    if (lane < 16) myid = neg[(size_t)n * KNEG + c * 16 + lane];

    // batch 16 row-gathers first: 16 outstanding global_load_b128 per wave (MLP vs HBM latency)
    v4f ev[16];
#pragma unroll
    for (int j = 0; j < 16; ++j) {
      int id = __shfl(myid, j, 32) - 1;
      id = id < 0 ? 0 : (id > maxidx ? maxidx : id);
      ev[j] = *(const v4f*)&table[(size_t)id * D_DIM + lane * 4];
    }
    // normalize each candidate, write f16 row to LDS
#pragma unroll
    for (int j = 0; j < 16; ++j) {
      float ss = wave_sum32(ev[j][0]*ev[j][0] + ev[j][1]*ev[j][1] +
                            ev[j][2]*ev[j][2] + ev[j][3]*ev[j][3]);
      float sc = 1.0f / fmaxf(sqrtf(ss), EPSN);
      v4h h;
#pragma unroll
      for (int k = 0; k < 4; ++k) h[k] = (_Float16)(ev[j][k] * sc);
      *(v4h*)&sE[wave][j * ESTRIDE + lane * 4] = h;
    }

    // 16 logits via chained WMMA: D = A(q-hat replicated 16xK) * B(16 candidates) + C
    v8f cm = {};
#pragma unroll
    for (int ck = 0; ck < 4; ++ck) {
      v16h b = *(const v16h*)&sE[wave][col * ESTRIDE + ck * 32 + bko];
      cm = __builtin_amdgcn_wmma_f32_16x16x32_f16(
          /*neg_a=*/false, Afrag[ck], /*neg_b=*/false, b,
          /*c_mod=*/(short)0, cm, /*reuse_a=*/false, /*reuse_b=*/false);
    }

    // C row M=0: lane j (0..15) holds dot(q-hat, e_j) in cm[0]
    if (lane < 16) {
      float x  = cm[0] * INV_TEMP;
      float mn = fmaxf(m, x);
      s = s * __expf(m - mn) + __expf(x - mn);
      m = mn;
    }
  }

  // merge (m, s) across lanes; lanes 16-31 carry (-inf, 0) = neutral
#pragma unroll
  for (int off = 16; off >= 1; off >>= 1) {
    float mo = __shfl_xor(m, off, 32);
    float so = __shfl_xor(s, off, 32);
    float mn = fmaxf(m, mo);
    s = s * __expf(m - mn) + so * __expf(mo - mn);
    m = mn;
  }
  // fold in the positive logit, finish loss = logsumexp - pos
  float mn   = fmaxf(m, pos_t);
  float S    = s * __expf(m - mn) + __expf(pos_t - mn);
  float loss = mn + __logf(S) - pos_t;
  if (lane == 0) lw[n] = loss * w;
}

__global__ void ssl_reduce_kernel(const float* __restrict__ lw,
                                  const float* __restrict__ sw,
                                  float* __restrict__ out, int n)
{
  __shared__ float s0[256];
  __shared__ float s1[256];
  float a = 0.0f, b = 0.0f;
  for (int i = threadIdx.x; i < n; i += 256) { a += lw[i]; b += sw[i]; }
  s0[threadIdx.x] = a; s1[threadIdx.x] = b;
  __syncthreads();
  for (int st = 128; st > 0; st >>= 1) {
    if (threadIdx.x < st) {
      s0[threadIdx.x] += s0[threadIdx.x + st];
      s1[threadIdx.x] += s1[threadIdx.x + st];
    }
    __syncthreads();
  }
  if (threadIdx.x == 0) out[0] = s0[0] / s1[0];
}

extern "C" void kernel_launch(void* const* d_in, const int* in_sizes, int n_in,
                              void* d_out, int out_size, void* d_ws, size_t ws_size,
                              hipStream_t stream)
{
  const float* out_emb = (const float*)d_in[0];
  const float* table   = (const float*)d_in[1];
  const float* sw      = (const float*)d_in[2];
  const int*   tgt     = (const int*)d_in[3];
  const int*   neg     = (const int*)d_in[4];

  const int n_rows     = in_sizes[0] / D_DIM;   // 4096
  const int table_rows = in_sizes[1] / D_DIM;   // 1,000,001

  float* lw = (float*)d_ws;                     // 4096 floats of scratch

  const int grid = (n_rows + ROWS_PER_BLK - 1) / ROWS_PER_BLK;
  ssl_row_kernel<<<grid, ROWS_PER_BLK * 32, 0, stream>>>(
      out_emb, table, sw, tgt, neg, lw, n_rows, table_rows);
  ssl_reduce_kernel<<<1, 256, 0, stream>>>(lw, sw, (float*)d_out, n_rows);
}
